// GQA_21981642621431
// MI455X (gfx1250) — compile-verified, matrix operand order checked by reading, not run on hardware
//
#include <hip/hip_runtime.h>
#include <hip/hip_bf16.h>

#define B_   4
#define N_   1024
#define Dm   4096
#define HD   128
#define HQ   32
#define HKV  4

typedef __attribute__((ext_vector_type(16))) __bf16 v16bf;
typedef __attribute__((ext_vector_type(8)))  float  v8f;

union FragAB { v16bf v; uint4 u[2]; };
union FragC  { v8f   v; float f[8]; };

__device__ inline unsigned short f2bf(float x) {
    unsigned u = __float_as_uint(x);
    unsigned r = u + 0x7FFFu + ((u >> 16) & 1u);   // round-to-nearest-even
    return (unsigned short)(r >> 16);
}

// ---------------------------------------------------------------------------
// Tensor Data Mover path. This toolchain's builtin takes 6 args:
//   (u32x4 group0, i32x8 group1, i32x4 group2, i32x4 group3, i32x8 extra, i32 cpol)
// ---------------------------------------------------------------------------
#if defined(__gfx1250__) && __has_builtin(__builtin_amdgcn_tensor_load_to_lds) && \
    __has_builtin(__builtin_amdgcn_s_wait_tensorcnt)
#define USE_TDM_PATH 1
#endif

#ifdef USE_TDM_PATH
typedef __attribute__((ext_vector_type(4))) unsigned int u32x4;
typedef __attribute__((ext_vector_type(8))) int          i32x8;
typedef __attribute__((ext_vector_type(4))) int          i32x4;

// DMA a [tile_h x tile_w] bf16 tile (row stride = row_stride elems) from global
// into LDS at byte offset lds_off, inserting 16B padding after every row
// (tile_w=64 elems = 32 DWORDs -> pad_interval code 4, pad_amount code 3),
// i.e. LDS row stride = 144B = 72 ushorts.
__device__ inline void tdm_load_tile(unsigned lds_off, const unsigned short* gptr,
                                     unsigned tile_w, unsigned tile_h,
                                     unsigned row_stride)
{
    unsigned long long ga = (unsigned long long)gptr;
    u32x4 g0;
    g0[0] = 1u;                                             // count=1, user desc
    g0[1] = lds_off;                                        // lds_addr
    g0[2] = (unsigned)(ga & 0xFFFFFFFFu);                   // global_addr[31:0]
    g0[3] = (unsigned)((ga >> 32) & 0x1FFFFFFu) | (2u << 30); // addr[56:32], type=2
    i32x8 g1;
    g1[0] = (int)((1u << 16)        // data_size = 2 bytes
                | (1u << 20)        // pad_enable
                | (4u << 22)        // pad_interval: 32 DWORDs
                | (3u << 25));      // pad_amount:   4 DWORDs (16B)
    unsigned td0 = row_stride;      // tensor_dim0 (>= tile_w, no OOB here)
    unsigned td1 = 0x3FFFFFu;       // tensor_dim1 (large, no OOB)
    g1[1] = (int)((td0 & 0xFFFFu) << 16);
    g1[2] = (int)(((td0 >> 16) & 0xFFFFu) | ((td1 & 0xFFFFu) << 16));
    g1[3] = (int)(((td1 >> 16) & 0xFFFFu) | ((tile_w & 0xFFFFu) << 16));
    g1[4] = (int)(tile_h & 0xFFFFu);                        // tile_dim1, tile_dim2=0
    g1[5] = (int)row_stride;                                // tensor_dim0_stride lo
    g1[6] = 0;
    g1[7] = 0;
    i32x4 z4 = {0, 0, 0, 0};
    i32x8 z8 = {0, 0, 0, 0, 0, 0, 0, 0};
    __builtin_amdgcn_tensor_load_to_lds(g0, g1, z4, z4, z8, 0);
}

__device__ inline unsigned lds_byte_off(const void* p) {
    return (unsigned)(unsigned long long)p;   // LDS aperture: addr[31:0] = offset
}
#endif

// ---------------------------------------------------------------------------
// f32 -> bf16 elementwise convert (for x)
// ---------------------------------------------------------------------------
__global__ __launch_bounds__(256) void conv_bf16(
    const float* __restrict__ s, unsigned short* __restrict__ d)
{
    int i = (blockIdx.x * 256 + threadIdx.x) * 8;
    float4 a = *(const float4*)(s + i);
    float4 b = *(const float4*)(s + i + 4);
    union { unsigned short h[8]; uint4 u; } t;
    t.h[0] = f2bf(a.x); t.h[1] = f2bf(a.y); t.h[2] = f2bf(a.z); t.h[3] = f2bf(a.w);
    t.h[4] = f2bf(b.x); t.h[5] = f2bf(b.y); t.h[6] = f2bf(b.z); t.h[7] = f2bf(b.w);
    *(uint4*)(d + i) = t.u;
}

// ---------------------------------------------------------------------------
// f32 [K,N] -> bf16 transposed [N,K] (LDS-tiled 32x32)
// ---------------------------------------------------------------------------
__global__ __launch_bounds__(256) void convtr_bf16(
    const float* __restrict__ src, unsigned short* __restrict__ dst, int K, int N)
{
    __shared__ float t[32][33];
    int kx = blockIdx.x * 32, ny = blockIdx.y * 32;
    int lx = threadIdx.x & 31, ly = threadIdx.x >> 5;    // 32 x 8
    #pragma unroll
    for (int i = 0; i < 32; i += 8)
        t[ly + i][lx] = src[(size_t)(kx + ly + i) * N + ny + lx];
    __syncthreads();
    #pragma unroll
    for (int i = 0; i < 32; i += 8)
        dst[(size_t)(ny + ly + i) * K + kx + lx] = f2bf(t[lx][ly + i]);
}

// ---------------------------------------------------------------------------
// C[M,N] = A[M,K](bf16) * Bt[N,K](bf16, pre-transposed) + bias, f32 out.
// Block: 256 thr = 8 waves (2 M x 4 N). Tile 128x64, BK=64, double-buffered.
// TDM path: wave 0 DMAs next tiles while all waves run WMMAs on current.
// ---------------------------------------------------------------------------
__global__ __launch_bounds__(256) void gemm_bf16(
    const unsigned short* __restrict__ A,
    const unsigned short* __restrict__ Bt,
    const float* __restrict__ bias, float* __restrict__ C,
    int M, int N, int K)
{
    __shared__ __align__(16) unsigned short As[2][128][72];  // 144B row stride
    __shared__ __align__(16) unsigned short Bs[2][64][72];

    const int tid   = threadIdx.x;
    const int wave  = tid >> 5, lane = tid & 31;
    const int lrow  = lane & 15, lhalf = lane >> 4;
    const int wm    = wave >> 2, wn = wave & 3;
    const int mbase = blockIdx.y * 128, nbase = blockIdx.x * 64;
    const int nk    = K >> 6;

    FragC acc[4];
    #pragma unroll
    for (int i = 0; i < 4; i++)
        #pragma unroll
        for (int r = 0; r < 8; r++) acc[i].f[r] = 0.0f;

#ifdef USE_TDM_PATH
    if (wave == 0) {
        tdm_load_tile(lds_byte_off(&As[0][0][0]), A  + (size_t)mbase * K, 64, 128, K);
        tdm_load_tile(lds_byte_off(&Bs[0][0][0]), Bt + (size_t)nbase * K, 64,  64, K);
    }
#endif

    for (int ki = 0; ki < nk; ki++) {
        const int cur = ki & 1;
        const int kb  = ki << 6;
#ifdef USE_TDM_PATH
        if (wave == 0) __builtin_amdgcn_s_wait_tensorcnt(0);
        __syncthreads();                 // cur buffer ready; back buffer free
        if (wave == 0 && ki + 1 < nk) {  // overlap next DMA with compute
            const int kb2 = kb + 64;
            tdm_load_tile(lds_byte_off(&As[cur ^ 1][0][0]),
                          A + (size_t)mbase * K + kb2, 64, 128, K);
            tdm_load_tile(lds_byte_off(&Bs[cur ^ 1][0][0]),
                          Bt + (size_t)nbase * K + kb2, 64, 64, K);
        }
#else
        {   // cooperative bf16 tile loads (no conversion needed)
            int r = tid >> 1, h = tid & 1;
            const uint4* ap = (const uint4*)(A + (size_t)(mbase + r) * K + kb + h * 32);
            uint4* dp = (uint4*)&As[cur][r][h * 32];
            dp[0] = ap[0]; dp[1] = ap[1]; dp[2] = ap[2]; dp[3] = ap[3];
        }
        {
            int r = tid >> 2, qd = tid & 3;
            const uint4* bp = (const uint4*)(Bt + (size_t)(nbase + r) * K + kb + qd * 16);
            uint4* dp = (uint4*)&Bs[cur][r][qd * 16];
            dp[0] = bp[0]; dp[1] = bp[1];
        }
        __syncthreads();
#endif
        #pragma unroll
        for (int kc = 0; kc < 2; kc++) {
            FragAB bfr;
            const uint4* bp = (const uint4*)&Bs[cur][wn * 16 + lrow][kc * 32];
            bfr.u[0] = bp[lhalf]; bfr.u[1] = bp[lhalf + 2];
            #pragma unroll
            for (int mt = 0; mt < 4; mt++) {
                FragAB a;
                const uint4* ap = (const uint4*)&As[cur][wm * 64 + mt * 16 + lrow][kc * 32];
                a.u[0] = ap[lhalf]; a.u[1] = ap[lhalf + 2];
                acc[mt].v = __builtin_amdgcn_wmma_f32_16x16x32_bf16(
                    false, a.v, false, bfr.v, (short)0, acc[mt].v, false, false);
            }
        }
#ifndef USE_TDM_PATH
        __syncthreads();
#endif
    }

    #pragma unroll
    for (int mt = 0; mt < 4; mt++) {
        int grow = mbase + wm * 64 + mt * 16 + lhalf * 8;
        int gcol = nbase + wn * 16 + lrow;
        float bv = bias[gcol];
        #pragma unroll
        for (int r = 0; r < 8; r++)
            C[(size_t)(grow + r) * N + gcol] = acc[mt].f[r] + bv;
    }
}

// ---------------------------------------------------------------------------
// RoPE + f32->bf16 + transpose [B*N, H*128] -> [B][H][N][128]
// ---------------------------------------------------------------------------
__global__ __launch_bounds__(256) void rope_pack(
    const float* __restrict__ src, unsigned short* __restrict__ dst,
    int H, int doRope)
{
    int idx = blockIdx.x * 256 + threadIdx.x;
    int j = idx & 63;
    int h = (idx >> 6) % H;
    int t = idx / (64 * H);
    int n = t % N_;
    int b = t / N_;
    size_t so = (size_t)(b * N_ + n) * (H * HD) + h * HD + 2 * j;
    float e = src[so], o = src[so + 1];
    if (doRope) {
        float freq = __expf(-((float)j * (1.0f / 64.0f)) * 9.210340372f); // ln(1e4)
        float ang = (float)n * freq;
        float s, c;
        __sincosf(ang, &s, &c);
        float e2 = e * c - o * s;
        o = o * c + e * s;
        e = e2;
    }
    size_t dof = ((size_t)(b * H + h) * N_ + n) * HD + 2 * j;
    dst[dof]     = f2bf(e);
    dst[dof + 1] = f2bf(o);
}

// ---------------------------------------------------------------------------
// Flash-style causal GQA attention. Grid: (N/128, B*HQ). 256 thr = 8 waves.
// Output bf16 [B*N, 4096] head-major, feeds the Wo GEMM directly.
// ---------------------------------------------------------------------------
__global__ __launch_bounds__(256) void attn_fwd(
    const unsigned short* __restrict__ qb,
    const unsigned short* __restrict__ kb,
    const unsigned short* __restrict__ vb,
    unsigned short* __restrict__ outb)
{
    __shared__ __align__(16) unsigned short vT[128][72];     // V^T [d][j]
    __shared__ __align__(16) unsigned short pb[8][16][72];   // per-wave P tile

    const int tid  = threadIdx.x;
    const int wave = tid >> 5, lane = tid & 31;
    const int lrow = lane & 15, lhalf = lane >> 4;
    const int b = blockIdx.y >> 5, hq = blockIdx.y & 31;
    const int hkv = hq & 3;
    const int nq0 = blockIdx.x * 128;
    const float scale = 0.08838834764831845f;     // 1/sqrt(128)

    const unsigned short* qhead = qb + (size_t)(b * HQ  + hq ) * N_ * HD;
    const unsigned short* khead = kb + (size_t)(b * HKV + hkv) * N_ * HD;
    const unsigned short* vhead = vb + (size_t)(b * HKV + hkv) * N_ * HD;

    FragAB qf[4];
    {
        int qrow = nq0 + wave * 16 + lrow;
        const uint4* qp = (const uint4*)(qhead + (size_t)qrow * HD);
        #pragma unroll
        for (int kc = 0; kc < 4; kc++) {
            qf[kc].u[0] = qp[kc * 4 + lhalf];
            qf[kc].u[1] = qp[kc * 4 + lhalf + 2];
        }
    }

    FragC o[8];
    #pragma unroll
    for (int dt = 0; dt < 8; dt++)
        #pragma unroll
        for (int r = 0; r < 8; r++) o[dt].f[r] = 0.0f;
    float rowmax[8], rowsum[8];
    #pragma unroll
    for (int r = 0; r < 8; r++) { rowmax[r] = -1e30f; rowsum[r] = 0.0f; }

    const int jblocks = (nq0 + 128) >> 6;
    for (int jb = 0; jb < jblocks; jb++) {
        int j0 = jb * 64;
        __syncthreads();
        {   // stage V block (64x128) transposed into LDS
            int j = tid >> 2, seg = tid & 3;
            const uint4* vp = (const uint4*)(vhead + (size_t)(j0 + j) * HD + seg * 32);
            union { uint4 u[4]; unsigned short h[32]; } tv;
            tv.u[0] = vp[0]; tv.u[1] = vp[1]; tv.u[2] = vp[2]; tv.u[3] = vp[3];
            #pragma unroll
            for (int i = 0; i < 32; i++) vT[seg * 32 + i][j] = tv.h[i];
        }
        __syncthreads();

        FragC S[4];
        #pragma unroll
        for (int nt = 0; nt < 4; nt++) {
            #pragma unroll
            for (int r = 0; r < 8; r++) S[nt].f[r] = 0.0f;
            int jrow = j0 + nt * 16 + lrow;
            const uint4* kp = (const uint4*)(khead + (size_t)jrow * HD);
            #pragma unroll
            for (int kc = 0; kc < 4; kc++) {
                FragAB kf;
                kf.u[0] = kp[kc * 4 + lhalf];
                kf.u[1] = kp[kc * 4 + lhalf + 2];
                S[nt].v = __builtin_amdgcn_wmma_f32_16x16x32_bf16(
                    false, qf[kc].v, false, kf.v, (short)0, S[nt].v, false, false);
            }
        }
        #pragma unroll
        for (int nt = 0; nt < 4; nt++)
            #pragma unroll
            for (int r = 0; r < 8; r++) {
                int row = nq0 + wave * 16 + r + lhalf * 8;
                int col = j0 + nt * 16 + lrow;
                float v = S[nt].f[r] * scale;
                S[nt].f[r] = (col <= row) ? v : -1e30f;
            }
        float nmax[8], corr[8], bsum[8];
        #pragma unroll
        for (int r = 0; r < 8; r++) {
            float m = fmaxf(fmaxf(S[0].f[r], S[1].f[r]), fmaxf(S[2].f[r], S[3].f[r]));
            #pragma unroll
            for (int d = 1; d < 16; d <<= 1) m = fmaxf(m, __shfl_xor(m, d, 32));
            nmax[r] = fmaxf(rowmax[r], m);
            corr[r] = __expf(rowmax[r] - nmax[r]);
            rowmax[r] = nmax[r];
            bsum[r] = 0.0f;
        }
        #pragma unroll
        for (int nt = 0; nt < 4; nt++)
            #pragma unroll
            for (int r = 0; r < 8; r++) {
                float p = __expf(S[nt].f[r] - nmax[r]);
                bsum[r] += p;
                pb[wave][r + lhalf * 8][nt * 16 + lrow] = f2bf(p);
            }
        #pragma unroll
        for (int r = 0; r < 8; r++) {
            float s = bsum[r];
            #pragma unroll
            for (int d = 1; d < 16; d <<= 1) s += __shfl_xor(s, d, 32);
            rowsum[r] = rowsum[r] * corr[r] + s;
        }
        #pragma unroll
        for (int dt = 0; dt < 8; dt++)
            #pragma unroll
            for (int r = 0; r < 8; r++) o[dt].f[r] *= corr[r];

        asm volatile("s_wait_dscnt 0" ::: "memory");  // P LDS writes -> reads

        FragAB pf[2];
        #pragma unroll
        for (int kc = 0; kc < 2; kc++) {
            const uint4* pp = (const uint4*)&pb[wave][lrow][kc * 32];
            pf[kc].u[0] = pp[lhalf];
            pf[kc].u[1] = pp[lhalf + 2];
        }
        #pragma unroll
        for (int dt = 0; dt < 8; dt++) {
            const uint4* vp4 = (const uint4*)&vT[dt * 16 + lrow][0];
            #pragma unroll
            for (int kc = 0; kc < 2; kc++) {
                FragAB vf;
                vf.u[0] = vp4[kc * 4 + lhalf];
                vf.u[1] = vp4[kc * 4 + lhalf + 2];
                o[dt].v = __builtin_amdgcn_wmma_f32_16x16x32_bf16(
                    false, pf[kc].v, false, vf.v, (short)0, o[dt].v, false, false);
            }
        }
    }

    #pragma unroll
    for (int r = 0; r < 8; r++) rowsum[r] = 1.0f / rowsum[r];
    #pragma unroll
    for (int dt = 0; dt < 8; dt++) {
        int gcol = hq * HD + dt * 16 + lrow;
        #pragma unroll
        for (int r = 0; r < 8; r++) {
            int grow = b * N_ + nq0 + wave * 16 + r + lhalf * 8;
            outb[(size_t)grow * Dm + gcol] = f2bf(o[dt].f[r] * rowsum[r]);
        }
    }
}

// ---------------------------------------------------------------------------
extern "C" void kernel_launch(void* const* d_in, const int* in_sizes, int n_in,
                              void* d_out, int out_size, void* d_ws, size_t ws_size,
                              hipStream_t stream)
{
    const float* x  = (const float*)d_in[0];
    const float* Wq = (const float*)d_in[1];
    const float* bq = (const float*)d_in[2];
    const float* Wk = (const float*)d_in[3];
    const float* bk = (const float*)d_in[4];
    const float* Wv = (const float*)d_in[5];
    const float* bv = (const float*)d_in[6];
    const float* Wo = (const float*)d_in[7];
    const float* bo = (const float*)d_in[8];
    float* out = (float*)d_out;

    const size_t MB = 1ull << 20;
    char* ws = (char*)d_ws;
    float* q32          = (float*)(ws);                    // 64 MB
    float* k32          = (float*)(ws + 64 * MB);          //  8 MB
    float* v32          = (float*)(ws + 72 * MB);          //  8 MB
    unsigned short* qb  = (unsigned short*)(ws + 80  * MB); // 32 MB
    unsigned short* kb  = (unsigned short*)(ws + 112 * MB); //  4 MB
    unsigned short* vb  = (unsigned short*)(ws + 116 * MB); //  4 MB
    unsigned short* xb  = (unsigned short*)(ws + 120 * MB); // 32 MB
    unsigned short* WqT = (unsigned short*)(ws + 152 * MB); // 32 MB
    unsigned short* WkT = (unsigned short*)(ws + 184 * MB); //  4 MB
    unsigned short* WvT = (unsigned short*)(ws + 188 * MB); //  4 MB
    unsigned short* WoT = (unsigned short*)(ws + 192 * MB); // 32 MB
    unsigned short* attnob = (unsigned short*)(ws);         // aliases dead q32

    const int M = B_ * N_;  // 4096
    dim3 blk(256);

    // one-time bf16 conversions (weights transposed to [N][K])
    conv_bf16<<<(M * Dm) / (8 * 256), blk, 0, stream>>>(x, xb);
    convtr_bf16<<<dim3(Dm / 32, Dm / 32),  blk, 0, stream>>>(Wq, WqT, Dm, Dm);
    convtr_bf16<<<dim3(Dm / 32, 512 / 32), blk, 0, stream>>>(Wk, WkT, Dm, 512);
    convtr_bf16<<<dim3(Dm / 32, 512 / 32), blk, 0, stream>>>(Wv, WvT, Dm, 512);
    convtr_bf16<<<dim3(Dm / 32, Dm / 32),  blk, 0, stream>>>(Wo, WoT, Dm, Dm);

    gemm_bf16<<<dim3(Dm / 64, M / 128),  blk, 0, stream>>>(xb, WqT, bq, q32, M, Dm, Dm);
    gemm_bf16<<<dim3(512 / 64, M / 128), blk, 0, stream>>>(xb, WkT, bk, k32, M, 512, Dm);
    gemm_bf16<<<dim3(512 / 64, M / 128), blk, 0, stream>>>(xb, WvT, bv, v32, M, 512, Dm);

    rope_pack<<<(B_ * N_ * HQ  * 64) / 256, blk, 0, stream>>>(q32, qb, HQ, 1);
    rope_pack<<<(B_ * N_ * HKV * 64) / 256, blk, 0, stream>>>(k32, kb, HKV, 1);
    rope_pack<<<(B_ * N_ * HKV * 64) / 256, blk, 0, stream>>>(v32, vb, HKV, 0);

    attn_fwd<<<dim3(N_ / 128, B_ * HQ), blk, 0, stream>>>(qb, kb, vb, attnob);

    gemm_bf16<<<dim3(Dm / 64, M / 128), blk, 0, stream>>>(attnob, WoT, bo, out, M, Dm, Dm);
}
